// GCNModel_73212012528270
// MI455X (gfx1250) — compile-verified, matrix operand order checked by reading, not run on hardware
//
#include <hip/hip_runtime.h>
#include <hip/hip_bf16.h>
#include <math.h>

#define NN 16384
#define HH 32

typedef __attribute__((ext_vector_type(16))) __bf16 v16bf;
typedef __attribute__((ext_vector_type(8)))  __bf16 v8bf;
typedef __attribute__((ext_vector_type(8)))  float  v8f;
typedef __attribute__((ext_vector_type(4)))  float  v4f;   // native vector for NT loads

// Swizzled element index for the WMMA B-operand layout of V_WMMA_F32_16X16X32_BF16.
// Storage: [kb][cg][lane][j], 16 contiguous bf16 per lane (two b128 loads).
// Lane half h covers K offsets {8h..8h+7, 16+8h..23+8h}; element j<8 -> K=j+8h,
// j>=8 -> K=16+(j-8)+8h.
__device__ __host__ __forceinline__ size_t swz_idx(int k, int c) {
  int kb   = k >> 5;
  int kr   = k & 31;
  int half = (kr >> 3) & 1;
  int j    = (kr & 7) | ((kr & 16) >> 1);
  int lane = half * 16 + (c & 15);
  int cg   = c >> 4;
  return ((((size_t)kb * 2 + cg) * 32 + lane) << 4) + j;
}

// Build X0 = inputs @ W0  ([N,1]x[1,32]) directly in swizzled bf16 layout.
__global__ void prep_x0(const float* __restrict__ in, const float* __restrict__ W0,
                        __bf16* __restrict__ x0) {
  int gid = blockIdx.x * blockDim.x + threadIdx.x;  // grid exactly covers N*H
  int k = gid >> 5;
  int c = gid & 31;
  x0[swz_idx(k, c)] = (__bf16)(in[k] * W0[c]);
}

// One GCN propagation pass: T = adj @ Xin (bf16 WMMA, f32 accum), then epilogue:
//   MODE 1: H = relu(T); Xout = swz_bf16(H @ W1)        (Wn = W1 [32x32])
//   MODE 2: H = relu(T); Xout = swz_bf16([H @ W2, 0..]) (Wn = W2 [32x1], col 0 only)
//   MODE 3: out = sigmoid(T[:,0])
// ASRC 0: read f32 adj (convert in regs).
// ASRC 1: read f32 adj, convert, AND stream the bf16 copy out to adj16.
// ASRC 2: read bf16 adj16 directly (half the bytes).
template <int MODE, int ASRC>
__global__ void __launch_bounds__(256)
gcn_pass(const float* __restrict__ adj, __bf16* __restrict__ adj16,
         const __bf16* __restrict__ xin, const float* __restrict__ Wn,
         __bf16* __restrict__ xout, float* __restrict__ out) {
  __shared__ float ldsP[8][512];  // per-wave partial 16x32 tiles
  __shared__ float ldsT[512];     // reduced (+relu) tile

  const int tid    = threadIdx.x;
  const int wave   = tid >> 5;
  const int lane   = tid & 31;
  const int half   = lane >> 4;   // K-half per ISA A/B-fragment layout
  const int nlo    = lane & 15;   // row (A) / column (B,C) within tile
  const int rowblk = blockIdx.x;
  const size_t row = (size_t)(rowblk * 16 + nlo);
  const float*  rp   = adj + row * NN;
  const __bf16* rp16 = (ASRC == 2) ? (const __bf16*)adj16 + row * NN : nullptr;
  __bf16*       wp16 = (ASRC == 1) ? adj16 + row * NN : nullptr;

  v8f c0 = {};  // output columns 0..15
  v8f c1 = {};  // output columns 16..31

  const int kbeg = wave * (NN / 8);
  const int kend = kbeg + (NN / 8);
#pragma unroll 4
  for (int kk = kbeg; kk < kend; kk += 32) {
    v16bf a;
    if (ASRC == 2) {
      // bf16 adj: two b128 loads per lane, non-temporal (pure stream).
      __builtin_prefetch(rp16 + kk + 1024, 0, 1);
      v8bf alo = __builtin_nontemporal_load((const v8bf*)(rp16 + kk + 8 * half));
      v8bf ahi = __builtin_nontemporal_load((const v8bf*)(rp16 + kk + 16 + 8 * half));
      a = __builtin_shufflevector(alo, ahi, 0, 1, 2, 3, 4, 5, 6, 7,
                                  8, 9, 10, 11, 12, 13, 14, 15);
    } else {
      // f32 adj: lanes L / L+16 take complementary K-halves of the same row so
      // every 128B line is fully consumed. Non-temporal: don't thrash L2.
      __builtin_prefetch(rp + kk + 512, 0, 1);
      const v4f a0 = __builtin_nontemporal_load((const v4f*)(rp + kk + 8 * half));
      const v4f a1 = __builtin_nontemporal_load((const v4f*)(rp + kk + 8 * half + 4));
      const v4f a2 = __builtin_nontemporal_load((const v4f*)(rp + kk + 16 + 8 * half));
      const v4f a3 = __builtin_nontemporal_load((const v4f*)(rp + kk + 16 + 8 * half + 4));
      v8bf alo, ahi;
      alo[0] = (__bf16)a0[0]; alo[1] = (__bf16)a0[1]; alo[2] = (__bf16)a0[2]; alo[3] = (__bf16)a0[3];
      alo[4] = (__bf16)a1[0]; alo[5] = (__bf16)a1[1]; alo[6] = (__bf16)a1[2]; alo[7] = (__bf16)a1[3];
      ahi[0] = (__bf16)a2[0]; ahi[1] = (__bf16)a2[1]; ahi[2] = (__bf16)a2[2]; ahi[3] = (__bf16)a2[3];
      ahi[4] = (__bf16)a3[0]; ahi[5] = (__bf16)a3[1]; ahi[6] = (__bf16)a3[2]; ahi[7] = (__bf16)a3[3];
      if (ASRC == 1) {
        // Stream the bf16 copy out once; passes 2/3 read half the bytes.
        __builtin_nontemporal_store(alo, (v8bf*)(wp16 + kk + 8 * half));
        __builtin_nontemporal_store(ahi, (v8bf*)(wp16 + kk + 16 + 8 * half));
      }
      a = __builtin_shufflevector(alo, ahi, 0, 1, 2, 3, 4, 5, 6, 7,
                                  8, 9, 10, 11, 12, 13, 14, 15);
    }

    // B fragments: pre-swizzled, 32 contiguous bytes per lane (1 MiB, L2-resident).
    const size_t kb = (size_t)(kk >> 5);
    v16bf b0 = *(const v16bf*)(xin + (((kb * 2 + 0) * 32 + lane) << 4));
    v16bf b1 = *(const v16bf*)(xin + (((kb * 2 + 1) * 32 + lane) << 4));

    c0 = __builtin_amdgcn_wmma_f32_16x16x32_bf16(false, a, false, b0, (short)0, c0, false, false);
    c1 = __builtin_amdgcn_wmma_f32_16x16x32_bf16(false, a, false, b1, (short)0, c1, false, false);
  }

  // C layout: element j of lane -> (M = j + 8*half, N = nlo [+16 for c1]).
#pragma unroll
  for (int j = 0; j < 8; ++j) {
    ldsP[wave][(j + 8 * half) * 32 + nlo]      = c0[j];
    ldsP[wave][(j + 8 * half) * 32 + 16 + nlo] = c1[j];
  }
  __syncthreads();

  for (int idx = tid; idx < 512; idx += 256) {
    float s = 0.f;
#pragma unroll
    for (int w = 0; w < 8; ++w) s += ldsP[w][idx];
    if (MODE != 3) s = fmaxf(s, 0.f);  // relu (not on final layer)
    ldsT[idx] = s;
  }
  __syncthreads();

  if (MODE == 1) {
    for (int idx = tid; idx < 512; idx += 256) {
      int m = idx >> 5, c = idx & 31;
      float z = 0.f;
#pragma unroll
      for (int h = 0; h < HH; ++h) z += ldsT[m * 32 + h] * Wn[h * 32 + c];
      xout[swz_idx(rowblk * 16 + m, c)] = (__bf16)z;
    }
  } else if (MODE == 2) {
    for (int idx = tid; idx < 512; idx += 256) {
      int m = idx >> 5, c = idx & 31;
      float z = 0.f;
      if (c == 0) {
#pragma unroll
        for (int h = 0; h < HH; ++h) z += ldsT[m * 32 + h] * Wn[h];
      }
      xout[swz_idx(rowblk * 16 + m, c)] = (__bf16)z;  // cols 1..31 zeroed
    }
  } else {  // MODE == 3
    if (tid < 16) {
      float t = ldsT[tid * 32];  // column 0 only
      out[rowblk * 16 + tid] = 1.f / (1.f + expf(-t));
    }
  }
}

extern "C" void kernel_launch(void* const* d_in, const int* in_sizes, int n_in,
                              void* d_out, int out_size, void* d_ws, size_t ws_size,
                              hipStream_t stream) {
  const float* inp = (const float*)d_in[0];  // [N,1]
  const float* adj = (const float*)d_in[1];  // [N,N] f32
  const float* W0  = (const float*)d_in[2];  // [1,32]
  const float* W1  = (const float*)d_in[3];  // [32,32]
  const float* W2  = (const float*)d_in[4];  // [32,1]
  float* out = (float*)d_out;

  // Workspace: three swizzled bf16 [N,32] operand buffers (1 MiB each),
  // then (if room) a 512 MiB bf16 adjacency cache at a 4 MiB offset.
  __bf16* x0 = (__bf16*)d_ws;
  __bf16* x1 = x0 + (size_t)NN * HH;
  __bf16* xv = x1 + (size_t)NN * HH;
  const size_t ADJ16_OFF = (size_t)4 << 20;
  __bf16* adj16 = (__bf16*)((char*)d_ws + ADJ16_OFF);
  const bool big_ws = ws_size >= ADJ16_OFF + (size_t)NN * NN * sizeof(__bf16);

  prep_x0<<<(NN * HH) / 256, 256, 0, stream>>>(inp, W0, x0);
  if (big_ws) {
    // 2.5 GiB total HBM traffic: 1R + 0.5W, then 0.5R + 0.5R.
    gcn_pass<1, 1><<<NN / 16, 256, 0, stream>>>(adj, adj16, x0, W1, x1, nullptr);
    gcn_pass<2, 2><<<NN / 16, 256, 0, stream>>>(adj, adj16, x1, W2, xv, nullptr);
    gcn_pass<3, 2><<<NN / 16, 256, 0, stream>>>(adj, adj16, xv, nullptr, nullptr, out);
  } else {
    // Fallback: 3x f32 reads (3 GiB).
    gcn_pass<1, 0><<<NN / 16, 256, 0, stream>>>(adj, nullptr, x0, W1, x1, nullptr);
    gcn_pass<2, 0><<<NN / 16, 256, 0, stream>>>(adj, nullptr, x1, W2, xv, nullptr);
    gcn_pass<3, 0><<<NN / 16, 256, 0, stream>>>(adj, nullptr, xv, nullptr, nullptr, out);
  }
}
